// GATEncoderLayer_17171279249943
// MI455X (gfx1250) — compile-verified, matrix operand order checked by reading, not run on hardware
//
#include <hip/hip_runtime.h>
#include <hip/hip_bf16.h>

// ---------------------------------------------------------------------------
// GAT layer: h = x@W (WMMA bf16, async-to-LDS staging), per-node attn dots,
// edge segment-softmax, head-mean-folded scatter aggregate, bias+relu.
// Sizes fixed by the reference.
// ---------------------------------------------------------------------------
#define NNODES 20000
#define NEDGES 320000
#define DIN    128
#define NHEADS 4
#define DOUT   128
#define HD     (NHEADS * DOUT)        // 512
#define ETOT   (NEDGES + NNODES)      // 340000
#define NEG_SLOPE 0.2f

#define LDS_STRIDE_U16 132            // 264 bytes: 8-aligned, bank-skewed

typedef __bf16 bf16x16 __attribute__((ext_vector_type(16)));
typedef float  f32x8   __attribute__((ext_vector_type(8)));

union Frag16 {
    bf16x16  v;
    unsigned u[8];
};

// monotone float <-> uint key (for atomicMax on floats incl. negatives)
__device__ __forceinline__ unsigned fkey(float f) {
    unsigned u = __float_as_uint(f);
    return (u & 0x80000000u) ? ~u : (u | 0x80000000u);
}
__device__ __forceinline__ float fkeyinv(unsigned k) {
    unsigned u = (k & 0x80000000u) ? (k ^ 0x80000000u) : ~k;
    return __uint_as_float(u);
}
__device__ __forceinline__ unsigned short f2bf(float f) {
    unsigned u = __float_as_uint(f);
    u += 0x7FFFu + ((u >> 16) & 1u);          // round-to-nearest-even
    return (unsigned short)(u >> 16);
}

// ---------------------------------------------------------------------------
// 0) init: zero acc [N,128], zero denom [N,H], maxkey = key(-inf)
// ---------------------------------------------------------------------------
__global__ void k_init(float* __restrict__ acc, float* __restrict__ denom,
                       unsigned* __restrict__ maxkey) {
    int idx = blockIdx.x * blockDim.x + threadIdx.x;
    const int total = NNODES * DOUT;
    for (; idx < total; idx += gridDim.x * blockDim.x) {
        acc[idx] = 0.0f;
        if (idx < NNODES * NHEADS) {
            denom[idx]  = 0.0f;
            maxkey[idx] = fkey(-__builtin_inff());
        }
    }
}

// ---------------------------------------------------------------------------
// 1a) convert x [N,128] fp32 -> xb bf16 (packed u32 pairs), row stride 256 B
// ---------------------------------------------------------------------------
__global__ void k_xcvt(const float* __restrict__ x, unsigned* __restrict__ xb32) {
    int idx = blockIdx.x * blockDim.x + threadIdx.x;   // N*64 pair slots
    if (idx >= NNODES * (DIN / 2)) return;
    float f0 = x[2 * idx];
    float f1 = x[2 * idx + 1];
    xb32[idx] = (unsigned)f2bf(f0) | ((unsigned)f2bf(f1) << 16);
}

// ---------------------------------------------------------------------------
// 1b) pack W [128,512] fp32 -> Wp[col][64] u32, u32 = bf16{W[2p][col], W[2p+1][col]}
//     (B-fragment pair order so GEMM B loads are plain u32 reads)
// ---------------------------------------------------------------------------
__global__ void k_wpack(const float* __restrict__ W, unsigned* __restrict__ Wp) {
    int idx = blockIdx.x * blockDim.x + threadIdx.x;   // 512*64 = 32768
    if (idx >= HD * (DIN / 2)) return;
    int col = idx >> 6;
    int p   = idx & 63;
    unsigned lo = f2bf(W[(2 * p)     * HD + col]);
    unsigned hi = f2bf(W[(2 * p + 1) * HD + col]);
    Wp[col * 64 + p] = lo | (hi << 16);
}

// ---------------------------------------------------------------------------
// 2) GEMM: h[N,512] = x[N,128] @ W[128,512] via v_wmma_f32_16x16x32_bf16
//    block = 256 threads (8 waves), tile = 128 rows x 64 cols, K=128.
//    A tile staged into LDS with global_load_async_to_lds_b64 (ASYNCcnt path).
// ---------------------------------------------------------------------------
__global__ __launch_bounds__(256) void k_gemm(const unsigned short* __restrict__ xb,
                                              const unsigned* __restrict__ Wp,
                                              float* __restrict__ h) {
    __shared__ __align__(16) unsigned short As[128 * LDS_STRIDE_U16];

    const int tid     = threadIdx.x;
    const int mBase   = blockIdx.x * 128;
    const int colBase = blockIdx.y * 64;

    // ---- async-stage bf16 A tile: 128 rows x 256 B, 8 B per lane per op ----
    {
        const unsigned ldsBase = (unsigned)(uintptr_t)&As[0];
        const unsigned long long gbase = (unsigned long long)(uintptr_t)xb;
        for (int i = 0; i < 16; ++i) {
            int idx  = tid + i * 256;              // 0..4095 8-byte chunks
            int row  = idx >> 5;                   // 32 chunks per row
            int ch   = idx & 31;
            int grow = mBase + row;
            if (grow >= NNODES) grow = 0;          // clamp; tail rows discarded at store
            unsigned long long ga = gbase + (unsigned long long)grow * (DIN * 2)
                                          + (unsigned)(ch * 8);
            unsigned la = ldsBase + (unsigned)(row * (LDS_STRIDE_U16 * 2) + ch * 8);
            asm volatile("global_load_async_to_lds_b64 %0, %1, off"
                         :: "v"(la), "v"(ga) : "memory");
        }
        asm volatile("s_wait_asynccnt 0x0" ::: "memory");
    }
    __syncthreads();

    const int w    = tid >> 5;                     // wave 0..7 -> row subtile
    const int lane = tid & 31;
    const int lrow = lane & 15;
    const int hi   = lane >> 4;
    const int arow = w * 16 + lrow;

    f32x8 c[4];
    for (int sub = 0; sub < 4; ++sub)
        for (int e = 0; e < 8; ++e) c[sub][e] = 0.0f;

    for (int s = 0; s < 4; ++s) {                  // K steps of 32
        Frag16 a;
        for (int g = 0; g < 8; ++g) {
            int ka = 32 * s + ((g < 4) ? 2 * g : 16 + 2 * (g - 4)) + hi * 8;
            a.u[g] = *(const unsigned*)&As[arow * LDS_STRIDE_U16 + ka];
        }
        for (int sub = 0; sub < 4; ++sub) {
            int col = colBase + sub * 16 + lrow;
            Frag16 b;
            for (int g = 0; g < 8; ++g) {
                int p = 16 * s + ((g < 4) ? g : 8 + (g - 4)) + hi * 4;
                b.u[g] = Wp[col * 64 + p];
            }
            c[sub] = __builtin_amdgcn_wmma_f32_16x16x32_bf16(
                false, a.v, false, b.v, (short)0, c[sub], false, false);
        }
    }

    for (int sub = 0; sub < 4; ++sub) {
        int col = colBase + sub * 16 + lrow;
        for (int r = 0; r < 8; ++r) {
            int grow = mBase + w * 16 + hi * 8 + r;
            if (grow < NNODES) h[(size_t)grow * HD + col] = c[sub][r];
        }
    }
}

// ---------------------------------------------------------------------------
// 3) per-node attention halves: a_src/a_dst [N,H] (one wave per node)
// ---------------------------------------------------------------------------
__global__ void k_attn(const float* __restrict__ h,
                       const float* __restrict__ att_src,
                       const float* __restrict__ att_dst,
                       float* __restrict__ a_src, float* __restrict__ a_dst) {
    int node = (blockIdx.x * blockDim.x + threadIdx.x) >> 5;
    int lane = threadIdx.x & 31;
    if (node >= NNODES) return;
    const float* hr = h + (size_t)node * HD;
    for (int head = 0; head < NHEADS; ++head) {
        float ss = 0.0f, sd = 0.0f;
        for (int j = 0; j < 4; ++j) {
            int k   = lane + j * 32;
            float v = hr[head * DOUT + k];
            ss += v * att_src[head * DOUT + k];
            sd += v * att_dst[head * DOUT + k];
        }
        for (int m = 16; m >= 1; m >>= 1) {
            ss += __shfl_xor(ss, m, 32);
            sd += __shfl_xor(sd, m, 32);
        }
        if (lane == 0) {
            a_src[node * NHEADS + head] = ss;
            a_dst[node * NHEADS + head] = sd;
        }
    }
}

__device__ __forceinline__ void edge_sd(const int* __restrict__ ei, int e,
                                        int& s, int& d) {
    if (e < NEDGES) { s = ei[e]; d = ei[NEDGES + e]; }
    else            { s = d = e - NEDGES; }              // self loops
}

// ---------------------------------------------------------------------------
// 4) segment max of leaky-relu(logits) per (dst, head) via uint-key atomicMax
// ---------------------------------------------------------------------------
__global__ void k_edge_max(const int* __restrict__ ei,
                           const float* __restrict__ a_src,
                           const float* __restrict__ a_dst,
                           unsigned* __restrict__ maxkey) {
    int e = blockIdx.x * blockDim.x + threadIdx.x;
    if (e >= ETOT) return;
    int s, d; edge_sd(ei, e, s, d);
    for (int head = 0; head < NHEADS; ++head) {
        float l = a_src[s * NHEADS + head] + a_dst[d * NHEADS + head];
        l = (l >= 0.0f) ? l : NEG_SLOPE * l;
        atomicMax(&maxkey[d * NHEADS + head], fkey(l));
    }
}

// ---------------------------------------------------------------------------
// 5) denom[d,h] = sum exp(logit - max)
// ---------------------------------------------------------------------------
__global__ void k_edge_denom(const int* __restrict__ ei,
                             const float* __restrict__ a_src,
                             const float* __restrict__ a_dst,
                             const unsigned* __restrict__ maxkey,
                             float* __restrict__ denom) {
    int e = blockIdx.x * blockDim.x + threadIdx.x;
    if (e >= ETOT) return;
    int s, d; edge_sd(ei, e, s, d);
    for (int head = 0; head < NHEADS; ++head) {
        float l = a_src[s * NHEADS + head] + a_dst[d * NHEADS + head];
        l = (l >= 0.0f) ? l : NEG_SLOPE * l;
        float m = fkeyinv(maxkey[d * NHEADS + head]);
        atomicAdd(&denom[d * NHEADS + head], __expf(l - m));
    }
}

// ---------------------------------------------------------------------------
// 6) scatter aggregate with head-mean folded in:
//    acc[d, t] += sum_h (alpha_h / H) * h[s, h, t]     (one block per edge)
// ---------------------------------------------------------------------------
__global__ __launch_bounds__(128) void k_aggregate(const int* __restrict__ ei,
                                                   const float* __restrict__ a_src,
                                                   const float* __restrict__ a_dst,
                                                   const unsigned* __restrict__ maxkey,
                                                   const float* __restrict__ denom,
                                                   const float* __restrict__ h,
                                                   float* __restrict__ acc) {
    __shared__ float sal[NHEADS];
    int e   = blockIdx.x;
    int tid = threadIdx.x;                               // 0..127
    int s, d; edge_sd(ei, e, s, d);
    if (tid < NHEADS) {
        float l = a_src[s * NHEADS + tid] + a_dst[d * NHEADS + tid];
        l = (l >= 0.0f) ? l : NEG_SLOPE * l;
        float m = fkeyinv(maxkey[d * NHEADS + tid]);
        sal[tid] = __expf(l - m) / denom[d * NHEADS + tid] * (1.0f / NHEADS);
    }
    __syncthreads();
    const float* hs = h + (size_t)s * HD;
    float f = sal[0] * hs[tid]
            + sal[1] * hs[DOUT + tid]
            + sal[2] * hs[2 * DOUT + tid]
            + sal[3] * hs[3 * DOUT + tid];
    atomicAdd(&acc[(size_t)d * DOUT + tid], f);
}

// ---------------------------------------------------------------------------
// 7) out = relu(acc + bias)
// ---------------------------------------------------------------------------
__global__ void k_finalize(const float* __restrict__ acc,
                           const float* __restrict__ bias,
                           float* __restrict__ out) {
    int idx = blockIdx.x * blockDim.x + threadIdx.x;
    if (idx >= NNODES * DOUT) return;
    out[idx] = fmaxf(acc[idx] + bias[idx & (DOUT - 1)], 0.0f);
}

// ---------------------------------------------------------------------------
extern "C" void kernel_launch(void* const* d_in, const int* in_sizes, int n_in,
                              void* d_out, int out_size, void* d_ws, size_t ws_size,
                              hipStream_t stream) {
    (void)in_sizes; (void)n_in; (void)out_size; (void)ws_size;

    const float* x       = (const float*)d_in[0];   // [N,128]
    const float* W       = (const float*)d_in[1];   // [128,512]
    const float* att_src = (const float*)d_in[2];   // [1,4,128]
    const float* att_dst = (const float*)d_in[3];   // [1,4,128]
    const float* bias    = (const float*)d_in[4];   // [128]
    const int*   ei      = (const int*)d_in[5];     // [2,E]
    float*       out     = (float*)d_out;           // [N,128]

    char* ws = (char*)d_ws;
    size_t cur = 0;
    auto carve = [&](size_t bytes) { void* p = ws + cur; cur += (bytes + 255) & ~(size_t)255; return p; };

    float*          h      = (float*)         carve((size_t)NNODES * HD * sizeof(float));   // 40.96 MB
    unsigned short* xb     = (unsigned short*) carve((size_t)NNODES * DIN * 2);             // 5.12 MB
    unsigned*       Wp     = (unsigned*)       carve((size_t)HD * (DIN / 2) * sizeof(unsigned));
    float*          a_src  = (float*)          carve((size_t)NNODES * NHEADS * sizeof(float));
    float*          a_dst  = (float*)          carve((size_t)NNODES * NHEADS * sizeof(float));
    unsigned*       maxkey = (unsigned*)       carve((size_t)NNODES * NHEADS * sizeof(unsigned));
    float*          denom  = (float*)          carve((size_t)NNODES * NHEADS * sizeof(float));
    float*          acc    = (float*)          carve((size_t)NNODES * DOUT * sizeof(float)); // 10.24 MB

    // 0) init accumulators
    k_init<<<10000, 256, 0, stream>>>(acc, denom, maxkey);
    // 1) convert x to bf16; pack W to bf16 fragment order
    k_xcvt<<<(NNODES * (DIN / 2) + 255) / 256, 256, 0, stream>>>(x, (unsigned*)xb);
    k_wpack<<<(HD * (DIN / 2) + 255) / 256, 256, 0, stream>>>(W, Wp);
    // 2) WMMA GEMM: h = x @ W  (async-to-LDS A staging)
    dim3 ggrid((NNODES + 127) / 128, HD / 64);
    k_gemm<<<ggrid, 256, 0, stream>>>(xb, Wp, h);
    // 3) attention halves
    k_attn<<<(NNODES + 7) / 8, 256, 0, stream>>>(h, att_src, att_dst, a_src, a_dst);
    // 4) segment max
    k_edge_max<<<(ETOT + 255) / 256, 256, 0, stream>>>(ei, a_src, a_dst, maxkey);
    // 5) softmax denominator
    k_edge_denom<<<(ETOT + 255) / 256, 256, 0, stream>>>(ei, a_src, a_dst, maxkey, denom);
    // 6) weighted scatter aggregate (head mean folded in)
    k_aggregate<<<ETOT, 128, 0, stream>>>(ei, a_src, a_dst, maxkey, denom, h, acc);
    // 7) bias + relu
    k_finalize<<<(NNODES * DOUT + 255) / 256, 256, 0, stream>>>(acc, bias, out);
}
